// Retina_47459388621061
// MI455X (gfx1250) — matrix operand order repaired and checked by reading, not compiled, not run on hardware
//
#include <hip/hip_runtime.h>
#include <hip/hip_bf16.h>
#include <math.h>

// ---- problem constants (from the reference) ----
#define RF        16
#define RF2       256      // RF*RF
#define KTAPS     20
#define T_FRAMES  400
#define H_IMG     180
#define W_IMG     320
#define PW        352      // W + 2*RF (padded frame width)
#define NWIN      381      // T - K + 1
#define ALPHA_C   50.0f

// ---- tiling ----
#define TILE_T       64    // output windows per workgroup
#define FRAMES_TILE  96    // frames of patches staged (6 M-tiles of 16)
#define FRAMES_USED  83    // TILE_T + KTAPS - 1 (rest gathered as zero)
#define YSTRIDE      33    // f32 Y row stride (padded: 32 taps + 1)

typedef __attribute__((ext_vector_type(16))) _Float16 v16h;
typedef __attribute__((ext_vector_type(8)))  _Float16 v8h;
typedef __attribute__((ext_vector_type(8)))  float    v8f;

__global__ __launch_bounds__(128)
void retina_wmma_kernel(const float* __restrict__ x,
                        const float* __restrict__ w,
                        const int*   __restrict__ rf_idx,
                        float*       __restrict__ out,
                        int n_neurons)
{
    // LDS: 48 KB patches (f16) + 16 KB filter (f16) + ~12.7 KB Y (f32)
    __shared__ _Float16 sPatch[FRAMES_TILE * RF2];
    __shared__ _Float16 sW[32 * RF2];
    __shared__ float    sY[FRAMES_TILE * YSTRIDE];

    const int tid  = threadIdx.x;
    const int lane = tid & 31;
    const int wv   = tid >> 5;          // wave id 0..3
    const int n    = blockIdx.y;        // neuron
    const int t0   = blockIdx.x * TILE_T;

    // Recover the patch origin in the zero-padded frame from rf_indices[n,0]
    // (the reference builds indices as local + v*PW + h, local[0] == 0).
    const int idx0 = rf_idx[n * RF2];
    const int v0   = idx0 / PW;
    const int h0   = idx0 - v0 * PW;

    // ---- stage filter (taps padded 20 -> 32 with zeros) as f16 ----
    for (int e = tid; e < 32 * RF2; e += 128) {
        const int tap = e >> 8, p = e & 255;
        const float val = (tap < KTAPS) ? w[tap * RF2 + p] : 0.0f;
        sW[e] = (_Float16)val;
    }

    // ---- gather patches for frames [t0, t0+FRAMES_TILE), f32 -> f16 ----
    // Reproduces jnp.pad zeros via boundary checks; coalesced 16-wide rows.
    for (int e = tid; e < FRAMES_TILE * RF2; e += 128) {
        const int fl = e >> 8, p = e & 255;
        const int t  = t0 + fl;
        float val = 0.0f;
        if (fl < FRAMES_USED && t < T_FRAMES) {
            const int r = v0 + (p >> 4) - RF;     // row in original frame
            const int c = h0 + (p & 15) - RF;     // col in original frame
            if ((unsigned)r < (unsigned)H_IMG && (unsigned)c < (unsigned)W_IMG)
                val = x[t * (H_IMG * W_IMG) + r * W_IMG + c];
        }
        sPatch[e] = (_Float16)val;
    }
    __syncthreads();

    // ---- WMMA: Y[frame, tap] = sum_p sPatch[frame, p] * sW[tap, p] ----
    // A fragment (16-bit 16x32): lanes 0-15 hold K 0-15, lanes 16-31
    // hold K 16-31, row M = lane&15, two f16x8 (16B) LDS loads.
    // B fragment (16-bit 32x16): lanes 0-15 hold K 0-15, lanes 16-31 hold
    // K 16-31, column N = lane&15 -> one contiguous 32B row slice of sW.
    const int row   = lane & 15;
    const int hi    = lane >> 4;
    const int abase = hi ? 16 : 0;
    const int bbase = hi ? 16 : 0;

    for (int m = wv; m < FRAMES_TILE / 16; m += 4) {
        v8f acc0 = {};   // taps 0..15
        v8f acc1 = {};   // taps 16..31 (only 16..19 meaningful)
        const _Float16* prow = &sPatch[(m * 16 + row) * RF2];
        #pragma unroll
        for (int q = 0; q < 8; ++q) {
            union { v16h v; v8h h[2]; } a;
            a.h[0] = *(const v8h*)(prow + q * 32 + abase);
            a.h[1] = *(const v8h*)(prow + q * 32 + abase + 8);
            const v16h b0 = *(const v16h*)(&sW[(row +  0) * RF2 + q * 32 + bbase]);
            const v16h b1 = *(const v16h*)(&sW[(row + 16) * RF2 + q * 32 + bbase]);
            acc0 = __builtin_amdgcn_wmma_f32_16x16x32_f16(
                       false, a.v, false, b0, (short)0, acc0, false, false);
            acc1 = __builtin_amdgcn_wmma_f32_16x16x32_f16(
                       false, a.v, false, b1, (short)0, acc1, false, false);
        }
        // D layout (f32 16x16): element (M = v + 8*hi, N = lane&15) in c[v]
        #pragma unroll
        for (int vv = 0; vv < 8; ++vv) {
            const int fr = m * 16 + vv + 8 * hi;
            sY[fr * YSTRIDE + row]      = acc0[vv];
            sY[fr * YSTRIDE + 16 + row] = acc1[vv];
        }
    }
    __syncthreads();

    // ---- temporal diagonal g[t] = sum_k Y[t+k, k], then 50*softplus ----
    if (tid < TILE_T) {
        const int t = t0 + tid;
        if (t < NWIN) {
            float g = 0.0f;
            #pragma unroll
            for (int k = 0; k < KTAPS; ++k)
                g += sY[(tid + k) * YSTRIDE + k];
            // ALPHA*softplus(BETA*(g-GAMMA)) with BETA=1, GAMMA=0, stable form
            const float sp = fmaxf(g, 0.0f) + log1pf(expf(-fabsf(g)));
            out[n * NWIN + t] = ALPHA_C * sp;
        }
    }
}

extern "C" void kernel_launch(void* const* d_in, const int* in_sizes, int n_in,
                              void* d_out, int out_size, void* d_ws, size_t ws_size,
                              hipStream_t stream)
{
    const float* x   = (const float*)d_in[0];   // (T, H, W) f32
    const float* w   = (const float*)d_in[1];   // (K*RF*RF,) f32
    const int*   idx = (const int*)d_in[2];     // (N, RF*RF) i32
    float*       out = (float*)d_out;           // (N, NWIN) f32

    const int n_neurons = in_sizes[2] / RF2;
    dim3 grid((NWIN + TILE_T - 1) / TILE_T, n_neurons, 1);
    retina_wmma_kernel<<<grid, 128, 0, stream>>>(x, w, idx, out, n_neurons);
}
